// WaveNetModelContinuos_60009283060250
// MI455X (gfx1250) — matrix operand order repaired and checked by reading, not compiled, not run on hardware
//
#include <hip/hip_runtime.h>
#include <hip/hip_bf16.h>

// ---------------------------------------------------------------------------
// WaveNet forward for MI455X (gfx1250, wave32, WMMA bf16 16x16x32, f32 accum)
//  - per-layer gates/residual: 2 time-tiles per wave, weights hoisted in VGPRs
//  - skip projection: single fused K=1600 GEMM over all 50 layers' g
// ---------------------------------------------------------------------------

typedef __attribute__((ext_vector_type(16))) __bf16 v16bf;
typedef __attribute__((ext_vector_type(8)))  __bf16 v8bf;
typedef __attribute__((ext_vector_type(8)))  float  v8f;

#define NB      8
#define TLEN    8192
#define OUTLEN  3077
#define HCAP    8224      // h/g time capacity: covers base+31 (+d<=512) reads
#define SCAP    3136      // 49*64: covers base+63 reads for 4-tile waves
#define NL      50
#define NT4     49        // ceil(3077/64)

__device__ __forceinline__ __bf16 f2bf(float f) {
  unsigned u = __builtin_bit_cast(unsigned, f);
  u += 0x7FFFu + ((u >> 16) & 1u);               // round to nearest even
  unsigned short s = (unsigned short)(u >> 16);
  return __builtin_bit_cast(__bf16, s);
}
__device__ __forceinline__ float bf2f(__bf16 b) {
  unsigned short s = __builtin_bit_cast(unsigned short, b);
  unsigned u = ((unsigned)s) << 16;
  return __builtin_bit_cast(float, u);
}
__device__ __forceinline__ v8f wmma_bf16(v16bf a, v16bf b, v8f c) {
  return __builtin_amdgcn_wmma_f32_16x16x32_bf16(false, a, false, b, (short)0, c,
                                                 false, false);
}

// ---- native CDNA5 transcendentals (V_TANH_F32 / V_EXP_F32) ----------------
__device__ __forceinline__ float fast_tanh(float x) {
#if __has_builtin(__builtin_amdgcn_tanhf)
  return __builtin_amdgcn_tanhf(x);
#elif __has_builtin(__builtin_amdgcn_tanh_f32)
  return __builtin_amdgcn_tanh_f32(x);
#else
  return tanhf(x);
#endif
}
__device__ __forceinline__ float fast_sigmoid(float x) {
  return __builtin_fmaf(0.5f, fast_tanh(0.5f * x), 0.5f);
}
__device__ __forceinline__ float fast_elu(float x) {
  return (x > 0.0f) ? x : (__expf(x) - 1.0f);
}

// K index inside a 16x32 bf16 A-fragment for (lane-half h, element j): contiguous k = 16h + j
__device__ __forceinline__ int a_kk(int h, int j) {
  return 16 * h + j;
}

// ------------------------------ setup --------------------------------------

// Pre-swizzle all weights into per-lane WMMA A-fragment order (bf16); also
// compute bsum[o] = sum_i b_skip[i][o] (folded into the fused skip GEMM).
__global__ void convert_weights_kernel(const float* __restrict__ w_sig,
                                       const float* __restrict__ w_tanh,
                                       const float* __restrict__ w_res,
                                       const float* __restrict__ w_skip,
                                       const float* __restrict__ w_post1,
                                       const float* __restrict__ b_skip,
                                       __bf16* __restrict__ Asig0, __bf16* __restrict__ Asig1,
                                       __bf16* __restrict__ Atanh0, __bf16* __restrict__ Atanh1,
                                       __bf16* __restrict__ Ares,
                                       __bf16* __restrict__ Askip,
                                       __bf16* __restrict__ Apost1,
                                       float* __restrict__ bsum) {
  long idx = (long)blockIdx.x * 256 + threadIdx.x;
  const long n32 = (long)NL * 1024;                 // one 32x32 matrix -> 2 tiles * 512
  if (idx < 5 * n32) {
    int  seg = (int)(idx / n32);
    long e   = idx % n32;
    int  i   = (int)(e / 1024);
    int  rem = (int)(e % 1024);
    int  mt = rem >> 9, lane = (rem >> 4) & 31, j = rem & 15;
    int  m = mt * 16 + (lane & 15);
    int  k = a_kk(lane >> 4, j);
    float v;
    if (seg == 0)      v = w_sig [(((long)i * 32 + m) * 32 + k) * 2 + 0];
    else if (seg == 1) v = w_sig [(((long)i * 32 + m) * 32 + k) * 2 + 1];
    else if (seg == 2) v = w_tanh[(((long)i * 32 + m) * 32 + k) * 2 + 0];
    else if (seg == 3) v = w_tanh[(((long)i * 32 + m) * 32 + k) * 2 + 1];
    else               v = w_res [((long)i * 32 + m) * 32 + k];
    __bf16* dst = (seg == 0) ? Asig0 : (seg == 1) ? Asig1 :
                  (seg == 2) ? Atanh0 : (seg == 3) ? Atanh1 : Ares;
    dst[e] = f2bf(v);
    return;
  }
  idx -= 5 * n32;
  const long nskip = (long)NL * 32 * 512;           // 50 layers * 32 M-tiles * 512
  if (idx < nskip) {
    int i = (int)(idx / 16384);
    int rem = (int)(idx % 16384);
    int mt = rem >> 9, lane = (rem >> 4) & 31, j = rem & 15;
    int m = mt * 16 + (lane & 15);
    int k = a_kk(lane >> 4, j);
    Askip[idx] = f2bf(w_skip[((long)i * 512 + m) * 32 + k]);
    return;
  }
  idx -= nskip;
  if (idx < 262144) {                               // post1: 16 K-tiles * 32 M-tiles * 512
    int j = (int)(idx & 15), lane = (int)((idx >> 4) & 31);
    int mt = (int)((idx >> 9) & 31), kt = (int)(idx >> 14);
    int m = mt * 16 + (lane & 15);
    int k = kt * 32 + a_kk(lane >> 4, j);
    Apost1[idx] = f2bf(w_post1[(long)m * 512 + k]);
    return;
  }
  idx -= 262144;
  if (idx < 512) {                                  // summed skip biases
    float s = 0.0f;
    for (int i = 0; i < NL; ++i) s += b_skip[(long)i * 512 + idx];
    bsum[idx] = s;
  }
}

// Input 1x1 conv: h[b,t,:32] = w_in(32x8) @ x[b,t,:8] + b_in; f32 + bf16 copies.
__global__ void in_conv_kernel(const float* __restrict__ x,
                               const float* __restrict__ w_in,
                               const float* __restrict__ b_in,
                               float* __restrict__ h32, __bf16* __restrict__ h16) {
  int idx = blockIdx.x * blockDim.x + threadIdx.x;
  if (idx >= NB * TLEN) return;
  int b = idx / TLEN, t = idx % TLEN;
  const float* xp = x + ((long)b * TLEN + t) * 8;
  float xv[8];
#pragma unroll
  for (int c = 0; c < 8; ++c) xv[c] = xp[c];
  float*  hp = h32 + ((long)b * HCAP + t) * 32;
  __bf16* hq = h16 + ((long)b * HCAP + t) * 32;
#pragma unroll
  for (int o = 0; o < 32; ++o) {
    float a = b_in[o];
#pragma unroll
    for (int c = 0; c < 8; ++c) a += w_in[o * 8 + c] * xv[c];
    hp[o] = a;
    hq[o] = f2bf(a);
  }
}

// ------------------------------ main layer ---------------------------------
// One wave per (batch, 32-timestep block): two 16-col tiles, A-frags hoisted.
__global__ __launch_bounds__(32)
void layer_kernel(const __bf16* __restrict__ h16_in, const float* __restrict__ h32_in,
                  __bf16* __restrict__ h16_out, float* __restrict__ h32_out,
                  __bf16* __restrict__ Gall,
                  const __bf16* __restrict__ Asig0, const __bf16* __restrict__ Asig1,
                  const __bf16* __restrict__ Atanh0, const __bf16* __restrict__ Atanh1,
                  const __bf16* __restrict__ Ares,
                  const float* __restrict__ b_sig, const float* __restrict__ b_tanh,
                  const float* __restrict__ b_res,
                  int layer, int d, int Lout, int nt2, int off) {
  __shared__ alignas(64) __bf16 sg[2][16 * 32];     // g tiles, (n, c) layout
  const int lane = threadIdx.x;
  const int n = lane & 15, half = lane >> 4;
  const int b = blockIdx.x / nt2, q = blockIdx.x % nt2;
  const int tbase = q * 32;

  // hoist the 10 weight A-fragments for this layer into VGPRs
  const v16bf* As0 = (const v16bf*)(Asig0  + (long)layer * 1024);
  const v16bf* As1 = (const v16bf*)(Asig1  + (long)layer * 1024);
  const v16bf* At0 = (const v16bf*)(Atanh0 + (long)layer * 1024);
  const v16bf* At1 = (const v16bf*)(Atanh1 + (long)layer * 1024);
  const v16bf* Ar  = (const v16bf*)(Ares   + (long)layer * 1024);
  v16bf aS00 = As0[lane],      aS01 = As1[lane];
  v16bf aS10 = As0[32 + lane], aS11 = As1[32 + lane];
  v16bf aT00 = At0[lane],      aT01 = At1[lane];
  v16bf aT10 = At0[32 + lane], aT11 = At1[32 + lane];
  v16bf aR0  = Ar[lane],       aR1  = Ar[32 + lane];

  // biases (D layout: vgpr r -> channel tile*16 + 8*half + r)
  const float* bs = b_sig  + (long)layer * 32;
  const float* bq = b_tanh + (long)layer * 32;
  const float* br = b_res  + (long)layer * 32;
  v8f vbs0, vbs1, vbt0, vbt1, vbr0, vbr1;
#pragma unroll
  for (int r = 0; r < 8; ++r) {
    vbs0[r] = bs[8 * half + r];      vbs1[r] = bs[16 + 8 * half + r];
    vbt0[r] = bq[8 * half + r];      vbt1[r] = bq[16 + 8 * half + r];
    vbr0[r] = br[8 * half + r];      vbr1[r] = br[16 + 8 * half + r];
  }

  const __bf16* hb   = h16_in + (long)b * HCAP * 32;
  const float*  h32b = h32_in + (long)b * HCAP * 32;
  float*        h32o = h32_out + (long)b * HCAP * 32;
  __bf16*       h16o = h16_out + (long)b * HCAP * 32;
  __bf16*       gb   = Gall + ((long)layer * NB + b) * SCAP * 32;

#pragma unroll
  for (int s = 0; s < 2; ++s) {
    const int t0 = tbase + s * 16;
    v16bf B0 = *(const v16bf*)(hb + (long)(t0 + n) * 32 + 16 * half);
    v16bf B1 = *(const v16bf*)(hb + (long)(t0 + n + d) * 32 + 16 * half);

    v8f cs0 = vbs0, cs1 = vbs1, ct0 = vbt0, ct1 = vbt1;
    cs0 = wmma_bf16(aS00, B0, cs0);
    cs0 = wmma_bf16(aS01, B1, cs0);
    cs1 = wmma_bf16(aS10, B0, cs1);
    cs1 = wmma_bf16(aS11, B1, cs1);
    ct0 = wmma_bf16(aT00, B0, ct0);
    ct0 = wmma_bf16(aT01, B1, ct0);
    ct1 = wmma_bf16(aT10, B0, ct1);
    ct1 = wmma_bf16(aT11, B1, ct1);

    v8bf p0, p1;
#pragma unroll
    for (int r = 0; r < 8; ++r) {
      float g0 = fast_sigmoid(cs0[r]) * fast_tanh(ct0[r]);
      float g1 = fast_sigmoid(cs1[r]) * fast_tanh(ct1[r]);
      p0[r] = f2bf(g0);
      p1[r] = f2bf(g1);
    }
    *(v8bf*)(&sg[s][n * 32 + 8 * half])      = p0;
    *(v8bf*)(&sg[s][n * 32 + 16 + 8 * half]) = p1;
    __syncthreads();

    // store g (skip input) for output-time tau = t - off in [0, OUTLEN)
    const int tau = t0 + n - off;
    if ((unsigned)tau < (unsigned)OUTLEN) {
      *(v8bf*)(gb + (long)tau * 32 + 8 * half)      = p0;
      *(v8bf*)(gb + (long)tau * 32 + 16 + 8 * half) = p1;
    }

    // residual 32x32 on g (re-read as B fragment)
    v16bf Bg = *(const v16bf*)(&sg[s][n * 32 + 16 * half]);
    v8f cr0 = vbr0, cr1 = vbr1;
    cr0 = wmma_bf16(aR0, Bg, cr0);
    cr1 = wmma_bf16(aR1, Bg, cr1);

    if ((t0 + n) < Lout) {
      const float* src = h32b + (long)(t0 + n + d) * 32;
      float*       dst = h32o + (long)(t0 + n) * 32;
      __bf16*      dst16 = h16o + (long)(t0 + n) * 32;
      v8bf hq0, hq1;
#pragma unroll
      for (int r = 0; r < 8; ++r) {
        float v = cr0[r] + src[8 * half + r];
        dst[8 * half + r] = v;
        hq0[r] = f2bf(v);
      }
#pragma unroll
      for (int r = 0; r < 8; ++r) {
        float v = cr1[r] + src[16 + 8 * half + r];
        dst[16 + 8 * half + r] = v;
        hq1[r] = f2bf(v);
      }
      *(v8bf*)(dst16 + 8 * half)      = hq0;
      *(v8bf*)(dst16 + 16 + 8 * half) = hq1;
    }
  }
}

// ----------------------- fused skip GEMM (K = 50*32) -----------------------
// ybf[b,tau,:512] = elu( sum_i w_skip[i] @ g_i[:,tau] + sum_i b_skip[i] )
__global__ __launch_bounds__(32)
void skip_fused_kernel(const __bf16* __restrict__ Gall, const __bf16* __restrict__ Askip,
                       const float* __restrict__ bsum, __bf16* __restrict__ ybf) {
  const int lane = threadIdx.x;
  const int n = lane & 15, half = lane >> 4;
  const int id = blockIdx.x;                        // ((b*NT4)+q)*32 + mt
  const int mt = id & 31;
  const int bt = id >> 5;
  const int b = bt / NT4, q = bt % NT4;
  const int t0 = q * 64;

  v8f c0, c1, c2, c3;
#pragma unroll
  for (int r = 0; r < 8; ++r) {
    float bv = bsum[mt * 16 + 8 * half + r];
    c0[r] = bv; c1[r] = bv; c2[r] = bv; c3[r] = bv;
  }
  for (int i = 0; i < NL; ++i) {
    v16bf A = ((const v16bf*)(Askip + (long)i * 16384))[mt * 32 + lane];
    const __bf16* gb = Gall + ((long)i * NB + b) * SCAP * 32;
    v16bf B0 = *(const v16bf*)(gb + (long)(t0 + n) * 32 + 16 * half);
    v16bf B1 = *(const v16bf*)(gb + (long)(t0 + 16 + n) * 32 + 16 * half);
    v16bf B2 = *(const v16bf*)(gb + (long)(t0 + 32 + n) * 32 + 16 * half);
    v16bf B3 = *(const v16bf*)(gb + (long)(t0 + 48 + n) * 32 + 16 * half);
    c0 = wmma_bf16(A, B0, c0);
    c1 = wmma_bf16(A, B1, c1);
    c2 = wmma_bf16(A, B2, c2);
    c3 = wmma_bf16(A, B3, c3);
  }
  v8f acc[4] = {c0, c1, c2, c3};
#pragma unroll
  for (int s = 0; s < 4; ++s) {
    int tau = t0 + s * 16 + n;
    if (tau < OUTLEN) {
      v8bf p;
#pragma unroll
      for (int r = 0; r < 8; ++r) p[r] = f2bf(fast_elu(acc[s][r]));
      *(v8bf*)(ybf + ((long)b * SCAP + tau) * 512 + mt * 16 + 8 * half) = p;
    }
  }
}

// y1 = elu(w_post1(512x512) @ ybf + b_post1); K=512 -> 16 WMMA steps, 4 tiles.
__global__ __launch_bounds__(32)
void post1_kernel(const __bf16* __restrict__ ybf, const __bf16* __restrict__ Apost1,
                  const float* __restrict__ b_post1, __bf16* __restrict__ y1) {
  const int lane = threadIdx.x;
  const int n = lane & 15, half = lane >> 4;
  const int id = blockIdx.x;                        // ((b*NT4)+q)*32 + mt
  const int mt = id & 31;
  const int bt = id >> 5;
  const int b = bt / NT4, q = bt % NT4;
  const int t0 = q * 64;

  v8f acc[4];
#pragma unroll
  for (int r = 0; r < 8; ++r) {
    float bv = b_post1[mt * 16 + 8 * half + r];
    acc[0][r] = bv; acc[1][r] = bv; acc[2][r] = bv; acc[3][r] = bv;
  }
  const __bf16* yb = ybf + (long)b * SCAP * 512;
  const v16bf* Ap = (const v16bf*)Apost1;
#pragma unroll
  for (int kt = 0; kt < 16; ++kt) {
    v16bf A = Ap[(kt * 32 + mt) * 32 + lane];
#pragma unroll
    for (int s = 0; s < 4; ++s) {
      v16bf B = *(const v16bf*)(yb + (long)(t0 + s * 16 + n) * 512 + kt * 32 + 16 * half);
      acc[s] = wmma_bf16(A, B, acc[s]);
    }
  }
#pragma unroll
  for (int s = 0; s < 4; ++s) {
    int tau = t0 + s * 16 + n;
    if (tau < OUTLEN) {
      v8bf p;
#pragma unroll
      for (int r = 0; r < 8; ++r) p[r] = f2bf(fast_elu(acc[s][r]));
      *(v8bf*)(y1 + ((long)b * SCAP + tau) * 512 + mt * 16 + 8 * half) = p;
    }
  }
}

// out[b,t] = b_post2 + sum_c w_post2[c] * y1[b,t,c]
__global__ void post2_kernel(const __bf16* __restrict__ y1, const float* __restrict__ w2,
                             const float* __restrict__ b2, float* __restrict__ out) {
  int idx = blockIdx.x * blockDim.x + threadIdx.x;
  if (idx >= NB * OUTLEN) return;
  int b = idx / OUTLEN, t = idx % OUTLEN;
  const __bf16* row = y1 + ((long)b * SCAP + t) * 512;
  float acc = b2[0];
#pragma unroll 8
  for (int c = 0; c < 512; ++c) acc += w2[c] * bf2f(row[c]);
  out[idx] = acc;
}

// -------------------------------- launcher ---------------------------------

extern "C" void kernel_launch(void* const* d_in, const int* in_sizes, int n_in,
                              void* d_out, int out_size, void* d_ws, size_t ws_size,
                              hipStream_t stream) {
  const float* x       = (const float*)d_in[0];
  const float* w_in    = (const float*)d_in[1];
  const float* b_in    = (const float*)d_in[2];
  const float* w_sig   = (const float*)d_in[3];
  const float* b_sig   = (const float*)d_in[4];
  const float* w_tanh  = (const float*)d_in[5];
  const float* b_tanh  = (const float*)d_in[6];
  const float* w_skip  = (const float*)d_in[7];
  const float* b_skip  = (const float*)d_in[8];
  const float* w_res   = (const float*)d_in[9];
  const float* b_res   = (const float*)d_in[10];
  const float* w_post1 = (const float*)d_in[11];
  const float* b_post1 = (const float*)d_in[12];
  const float* w_post2 = (const float*)d_in[13];
  const float* b_post2 = (const float*)d_in[14];
  float* out = (float*)d_out;

  // ---- workspace layout (all chunks 256B aligned by construction) ----
  char* p = (char*)d_ws;
  const size_t SZ32    = (size_t)NL * 1024 * 2;            // 102,400 per family
  const size_t SZSKIPW = (size_t)NL * 16384 * 2;           // 1,638,400
  const size_t SZP1W   = (size_t)262144 * 2;               // 524,288
  const size_t SZBSUM  = 2048;                             // 512 f32
  const size_t SZGALL  = (size_t)NL * NB * SCAP * 32 * 2;  // 80,281,600
  const size_t SZH32   = (size_t)NB * HCAP * 32 * 4;       // 8,421,376
  const size_t SZH16   = (size_t)NB * HCAP * 32 * 2;       // 4,210,688
  const size_t SZYBF   = (size_t)NB * SCAP * 512 * 2;      // 25,690,112

  __bf16* Asig0  = (__bf16*)p; p += SZ32;
  __bf16* Asig1  = (__bf16*)p; p += SZ32;
  __bf16* Atanh0 = (__bf16*)p; p += SZ32;
  __bf16* Atanh1 = (__bf16*)p; p += SZ32;
  __bf16* Ares   = (__bf16*)p; p += SZ32;
  __bf16* Askip  = (__bf16*)p; p += SZSKIPW;
  __bf16* Apost1 = (__bf16*)p; p += SZP1W;
  float*  bsum   = (float*)p;  p += SZBSUM;
  __bf16* Gall   = (__bf16*)p; p += SZGALL;
  char* pool = p;                       // h ping-pong; later aliased by ybf
  float*  h32a = (float*)(pool);
  float*  h32b = (float*)(pool + SZH32);
  __bf16* h16a = (__bf16*)(pool + 2 * SZH32);
  __bf16* h16b = (__bf16*)(pool + 2 * SZH32 + SZH16);
  __bf16* ybf  = (__bf16*)pool;         // pool covers max(2*SZH32+2*SZH16, SZYBF)
  __bf16* y1   = (__bf16*)Gall;         // G dead once skip_fused has run
  (void)ws_size; (void)in_sizes; (void)n_in; (void)out_size;
  // total ~= 2.7 MB weights + 80.3 MB G + 25.7 MB pool ~= 108.7 MB

  // 1) weight swizzle/convert + summed skip bias
  {
    long total = 5L * NL * 1024 + (long)NL * 32 * 512 + 262144 + 512;
    convert_weights_kernel<<<(unsigned)((total + 255) / 256), 256, 0, stream>>>(
        w_sig, w_tanh, w_res, w_skip, w_post1, b_skip,
        Asig0, Asig1, Atanh0, Atanh1, Ares, Askip, Apost1, bsum);
  }
  // 2) input 1x1 conv
  in_conv_kernel<<<(NB * TLEN + 255) / 256, 256, 0, stream>>>(x, w_in, b_in, h32a, h16a);

  // 3) 50 dilated layers (stream-serialized; ping-pong h; g stashed into Gall)
  int L = TLEN;
  int par = 0;
  for (int i = 0; i < NL; ++i) {
    int d = 1 << (i % 10);
    int Lout = L - d;
    int nt2 = (Lout + 31) / 32;
    int off = Lout - OUTLEN;
    const __bf16* h16i = par ? h16b : h16a;
    const float*  h32i = par ? h32b : h32a;
    __bf16* h16o = par ? h16a : h16b;
    float*  h32o = par ? h32a : h32b;
    layer_kernel<<<NB * nt2, 32, 0, stream>>>(h16i, h32i, h16o, h32o, Gall,
                                              Asig0, Asig1, Atanh0, Atanh1, Ares,
                                              b_sig, b_tanh, b_res, i, d, Lout, nt2, off);
    L = Lout;
    par ^= 1;
  }

  // 4) fused skip GEMM over all layers (K=1600) + elu -> bf16
  skip_fused_kernel<<<NB * NT4 * 32, 32, 0, stream>>>(Gall, Askip, bsum, ybf);
  // 5) post1 GEMM (512x512) + elu
  post1_kernel<<<NB * NT4 * 32, 32, 0, stream>>>(ybf, Apost1, b_post1, y1);
  // 6) final 1-channel projection
  post2_kernel<<<(NB * OUTLEN + 255) / 256, 256, 0, stream>>>(y1, w_post2, b_post2, out);
}
